// get_model_88038239633521
// MI455X (gfx1250) — compile-verified, matrix operand order checked by reading, not run on hardware
//
#include <hip/hip_runtime.h>
#include <math.h>

// ---------------------------------------------------------------------------
// PointNet++ MSG forward for MI455X (gfx1250, wave32).
// MLP layers -> bf16 WMMA GEMMs (v_wmma_f32_16x16x32_bf16, f32 accumulate).
// Activations stored as bf16 (K padded to 32), weights pre-packed in fragment
// order (N-tiles padded to multiple of 4). Branch-free, double-buffered K
// loop with pointer-bump addressing and a peeled last chunk:
//   steady state = 10x global_load_b128 + 4x v_wmma + pointer adds.
// ---------------------------------------------------------------------------

#define BATCH 8
#define NPTS  1024

typedef __bf16 bf16_t;
typedef __attribute__((ext_vector_type(16))) bf16_t v16bf;
typedef __attribute__((ext_vector_type(8)))  float  v8f;

struct B32x2 { uint4 lo, hi; };

__device__ __forceinline__ bf16_t f2bf(float x) {
  unsigned u = __builtin_bit_cast(unsigned, x);
  u += 0x7FFFu + ((u >> 16) & 1u);            // round-to-nearest-even
  unsigned short h = (unsigned short)(u >> 16);
  return __builtin_bit_cast(bf16_t, h);
}

__device__ __forceinline__ float bf2f(bf16_t h) {
  unsigned u = ((unsigned)__builtin_bit_cast(unsigned short, h)) << 16;
  return __builtin_bit_cast(float, u);
}

// ---------------------------------------------------------------------------
// (B,6,N) -> coords (B,N,3), normals (B,N,3)   (f32, used by FPS/ball query)
// ---------------------------------------------------------------------------
__global__ void split_xyz_kernel(const float* __restrict__ x,
                                 float* __restrict__ coords,
                                 float* __restrict__ norms, int n) {
  int t = blockIdx.x * blockDim.x + threadIdx.x;
  if (t >= BATCH * n) return;
  int b = t / n, i = t % n;
  const float* xb = x + (size_t)b * 6 * n;
  float* c = coords + (size_t)t * 3;
  float* m = norms  + (size_t)t * 3;
  c[0] = xb[0 * n + i]; c[1] = xb[1 * n + i]; c[2] = xb[2 * n + i];
  m[0] = xb[3 * n + i]; m[1] = xb[4 * n + i]; m[2] = xb[5 * n + i];
}

// ---------------------------------------------------------------------------
// Farthest point sampling: one workgroup per batch, distances in LDS.
// ---------------------------------------------------------------------------
__global__ void __launch_bounds__(256)
fps_kernel(const float* __restrict__ xyz, int n, int npoint,
           int* __restrict__ outIdx) {
  __shared__ float dist[NPTS];
  __shared__ float rv[256];
  __shared__ int   ri[256];
  int b = blockIdx.x;
  const float* p = xyz + (size_t)b * n * 3;
  for (int i = threadIdx.x; i < n; i += blockDim.x) dist[i] = 1e10f;
  __syncthreads();
  int farthest = 0;
  for (int s = 0; s < npoint; ++s) {
    if (threadIdx.x == 0) outIdx[b * npoint + s] = farthest;
    float cx = p[farthest * 3 + 0];
    float cy = p[farthest * 3 + 1];
    float cz = p[farthest * 3 + 2];
    float best = -1.0f; int bi = 0;
    for (int i = threadIdx.x; i < n; i += blockDim.x) {
      float dx = p[i * 3 + 0] - cx;
      float dy = p[i * 3 + 1] - cy;
      float dz = p[i * 3 + 2] - cz;
      float d  = dx * dx + dy * dy + dz * dz;
      float nd = fminf(dist[i], d);
      dist[i] = nd;
      if (nd > best) { best = nd; bi = i; }
    }
    rv[threadIdx.x] = best; ri[threadIdx.x] = bi;
    __syncthreads();
    for (int o = 128; o > 0; o >>= 1) {
      if (threadIdx.x < o) {
        if (rv[threadIdx.x + o] > rv[threadIdx.x] ||
            (rv[threadIdx.x + o] == rv[threadIdx.x] &&
             ri[threadIdx.x + o] < ri[threadIdx.x])) {
          rv[threadIdx.x] = rv[threadIdx.x + o];
          ri[threadIdx.x] = ri[threadIdx.x + o];
        }
      }
      __syncthreads();
    }
    farthest = ri[0];
    __syncthreads();
  }
}

// gather: dst[b,s,:] = src[b, idx[b,s], :]
__global__ void gather_xyz_kernel(const float* __restrict__ src,
                                  const int* __restrict__ idx,
                                  float* __restrict__ dst, int n, int S) {
  int t = blockIdx.x * blockDim.x + threadIdx.x;
  if (t >= BATCH * S) return;
  int b = t / S;
  int id = idx[t];
  const float* s = src + ((size_t)b * n + id) * 3;
  float* d = dst + (size_t)t * 3;
  d[0] = s[0]; d[1] = s[1]; d[2] = s[2];
}

// ---------------------------------------------------------------------------
// Ball query (reference semantics: first `ns` in-radius indices in index
// order; pad with first hit, or n-1 if none).
// ---------------------------------------------------------------------------
__global__ void query_ball_kernel(const float* __restrict__ xyz,
                                  const float* __restrict__ nxyz,
                                  int n, int S, int ns, float r2,
                                  int* __restrict__ out) {
  int t = blockIdx.x * blockDim.x + threadIdx.x;
  if (t >= BATCH * S) return;
  int b = t / S;
  const float* p = xyz + (size_t)b * n * 3;
  float cx = nxyz[t * 3 + 0], cy = nxyz[t * 3 + 1], cz = nxyz[t * 3 + 2];
  int* o = out + (size_t)t * ns;
  int cnt = 0, first = -1;
  for (int i = 0; i < n && cnt < ns; ++i) {
    float dx = p[i * 3 + 0] - cx;
    float dy = p[i * 3 + 1] - cy;
    float dz = p[i * 3 + 2] - cz;
    if (dx * dx + dy * dy + dz * dz <= r2) {
      if (first < 0) first = i;
      o[cnt++] = i;
    }
  }
  if (first < 0) first = n - 1;
  for (; cnt < ns; ++cnt) o[cnt] = first;
}

// grouped row (bf16): [pts[b,idx,:] | xyz[b,idx]-new_xyz[b,s] | zero pad]
__global__ void group_kernel(const float* __restrict__ pts, int C,
                             const float* __restrict__ xyz,
                             const float* __restrict__ nxyz,
                             const int* __restrict__ idx,
                             int S, int ns, int n, int ld,
                             bf16_t* __restrict__ out, int total) {
  int t = blockIdx.x * blockDim.x + threadIdx.x;
  if (t >= total) return;
  int s = (t / ns) % S;
  int b =  t / (ns * S);
  int id = idx[t];
  const float* ps = pts + ((size_t)b * n + id) * C;
  bf16_t* o = out + (size_t)t * ld;
  for (int c = 0; c < C; ++c) o[c] = f2bf(ps[c]);
  const float* px = xyz + ((size_t)b * n + id) * 3;
  const float* cx = nxyz + ((size_t)(b * S + s)) * 3;
  o[C + 0] = f2bf(px[0] - cx[0]);
  o[C + 1] = f2bf(px[1] - cx[1]);
  o[C + 2] = f2bf(px[2] - cx[2]);
  for (int c = C + 3; c < ld; ++c) o[c] = f2bf(0.0f);
}

// ---------------------------------------------------------------------------
// Weight pack: W[Kreal,N] f32 -> fragment-ordered bf16
//   out[((nt*KC + kc)*32 + lane)*16 + j]; NTpad tiles (zero beyond N).
// ---------------------------------------------------------------------------
__global__ void pack_w_kernel(const float* __restrict__ W, int Kreal, int N,
                              int KC, int NTpad, bf16_t* __restrict__ out) {
  int t = blockIdx.x * blockDim.x + threadIdx.x;
  int total = NTpad * KC * 32;
  if (t >= total) return;
  int lane = t & 31;
  int kc   = (t >> 5) % KC;
  int nt   = t / (32 * KC);
  int half = lane >> 4, lm = lane & 15;
  int col = nt * 16 + lm;
  bf16_t* o = out + (size_t)t * 16;
#pragma unroll
  for (int j = 0; j < 16; ++j) {
    int v = j >> 1;
    int kk = kc * 32 + ((v < 4) ? (2 * v) : (16 + 2 * (v - 4))) + 8 * half + (j & 1);
    float w = (col < N && kk < Kreal) ? W[(size_t)kk * N + col] : 0.0f;
    o[j] = f2bf(w);
  }
}

// ---------------------------------------------------------------------------
// bf16 WMMA GEMM: out[M,N] = act(bn(A @ W + bias))
// A: bf16 row-major, lda mult of 32. 8 waves/block; wave = 16 rows x 64 cols
// (4 accumulators, all 4 N-tiles always computed — pack buffer is padded).
// Pointer-bump double-buffered K loop, peeled last chunk.
// ---------------------------------------------------------------------------
__global__ void __launch_bounds__(256)
wmma_gemm_bf16(const bf16_t* __restrict__ A, int lda,
               const v16bf* __restrict__ Wp,
               const float* __restrict__ bias, const float* __restrict__ gam,
               const float* __restrict__ beta,
               void* __restrict__ out, int ldo,
               int Mcompute, int Mstore, int KC, int N,
               int fuse, int outF32) {
  const int lane = threadIdx.x & 31;
  const int wid  = threadIdx.x >> 5;
  const int half = lane >> 4;
  const int lm   = lane & 15;
  const int m0 = (blockIdx.x * 8 + wid) * 16;
  if (m0 >= Mcompute) return;            // uniform per-wave
  const int ntBase = blockIdx.y * 4;

  const uint4*  ap = (const uint4*)(A + (size_t)(m0 + lm) * lda) + half;
  const v16bf* p0 = Wp + ((size_t)(ntBase + 0) * KC) * 32 + lane;
  const v16bf* p1 = p0 + (size_t)KC * 32;
  const v16bf* p2 = p1 + (size_t)KC * 32;
  const v16bf* p3 = p2 + (size_t)KC * 32;

  v8f acc0 = {0,0,0,0,0,0,0,0};
  v8f acc1 = acc0, acc2 = acc0, acc3 = acc0;

  // prologue: fragments for kc = 0
  uint4  aLo = ap[0];
  uint4  aHi = ap[2];
  v16bf  b0 = p0[0], b1 = p1[0], b2 = p2[0], b3 = p3[0];

  for (int kc = 0; kc < KC - 1; ++kc) {
    B32x2 apair; apair.lo = aLo; apair.hi = aHi;
    v16bf af = __builtin_bit_cast(v16bf, apair);
    v16bf c0 = b0, c1 = b1, c2 = b2, c3 = b3;

    // advance and prefetch next chunk (pure pointer bumps)
    ap += 4; p0 += 32; p1 += 32; p2 += 32; p3 += 32;
    aLo = ap[0];
    aHi = ap[2];
    b0 = p0[0]; b1 = p1[0]; b2 = p2[0]; b3 = p3[0];

    acc0 = __builtin_amdgcn_wmma_f32_16x16x32_bf16(false, af, false, c0,
                                                   (short)0, acc0, false, false);
    acc1 = __builtin_amdgcn_wmma_f32_16x16x32_bf16(false, af, false, c1,
                                                   (short)0, acc1, false, false);
    acc2 = __builtin_amdgcn_wmma_f32_16x16x32_bf16(false, af, false, c2,
                                                   (short)0, acc2, false, false);
    acc3 = __builtin_amdgcn_wmma_f32_16x16x32_bf16(false, af, false, c3,
                                                   (short)0, acc3, false, false);
  }
  {
    // peeled final chunk
    B32x2 apair; apair.lo = aLo; apair.hi = aHi;
    v16bf af = __builtin_bit_cast(v16bf, apair);
    acc0 = __builtin_amdgcn_wmma_f32_16x16x32_bf16(false, af, false, b0,
                                                   (short)0, acc0, false, false);
    acc1 = __builtin_amdgcn_wmma_f32_16x16x32_bf16(false, af, false, b1,
                                                   (short)0, acc1, false, false);
    acc2 = __builtin_amdgcn_wmma_f32_16x16x32_bf16(false, af, false, b2,
                                                   (short)0, acc2, false, false);
    acc3 = __builtin_amdgcn_wmma_f32_16x16x32_bf16(false, af, false, b3,
                                                   (short)0, acc3, false, false);
  }

  const float inv = 0.9999950000374997f;  // 1/sqrt(1+1e-5)
  v8f accs[4] = {acc0, acc1, acc2, acc3};
#pragma unroll
  for (int t = 0; t < 4; ++t) {
    int n = (ntBase + t) * 16 + lm;
    if (n >= N) continue;
#pragma unroll
    for (int r = 0; r < 8; ++r) {
      int m = m0 + r + 8 * half;           // C/D layout: VGPR r -> M row
      if (m < Mstore) {
        float v = accs[t][r] + bias[n];
        if (fuse) {
          v = v * inv * gam[n] + beta[n];
          v = fmaxf(v, 0.0f);
        }
        if (outF32) ((float*)out)[(size_t)m * ldo + n] = v;
        else        ((bf16_t*)out)[(size_t)m * ldo + n] = f2bf(v);
      }
    }
  }
}

// max over `ns` axis of bf16 in[(G,ns) x ldin] -> f32 out[g, coff+c]
__global__ void maxpool_kernel(const bf16_t* __restrict__ in, int ldin,
                               float* __restrict__ out,
                               int G, int ns, int C, int Ctot, int coff) {
  int t = blockIdx.x * blockDim.x + threadIdx.x;
  if (t >= G * C) return;
  int g = t / C, c = t % C;
  const bf16_t* p = in + ((size_t)g * ns) * ldin + c;
  float m = -1e30f;
  for (int j = 0; j < ns; ++j) m = fmaxf(m, bf2f(p[(size_t)j * ldin]));
  out[(size_t)g * Ctot + coff + c] = m;
}

// rows (B*128, 672) bf16 = [l2_xyz(3) | l2_pts(640) | zero pad]
__global__ void sa3_concat_kernel(const float* __restrict__ xyz,
                                  const float* __restrict__ pts,
                                  bf16_t* __restrict__ out, int ld) {
  int t = blockIdx.x * blockDim.x + threadIdx.x;
  if (t >= BATCH * 128) return;
  bf16_t* o = out + (size_t)t * ld;
  o[0] = f2bf(xyz[t * 3 + 0]);
  o[1] = f2bf(xyz[t * 3 + 1]);
  o[2] = f2bf(xyz[t * 3 + 2]);
  const float* p = pts + (size_t)t * 640;
  for (int c = 0; c < 640; ++c) o[3 + c] = f2bf(p[c]);
  for (int c = 643; c < ld; ++c) o[c] = f2bf(0.0f);
}

__global__ void f32_to_bf16_kernel(const float* __restrict__ in,
                                   bf16_t* __restrict__ out, int n) {
  int t = blockIdx.x * blockDim.x + threadIdx.x;
  if (t < n) out[t] = f2bf(in[t]);
}

__global__ void log_softmax_kernel(const float* __restrict__ in,
                                   float* __restrict__ out, int R, int C) {
  int r = blockIdx.x * blockDim.x + threadIdx.x;
  if (r >= R) return;
  const float* x = in + (size_t)r * C;
  float mx = -1e30f;
  for (int c = 0; c < C; ++c) mx = fmaxf(mx, x[c]);
  float s = 0.0f;
  for (int c = 0; c < C; ++c) s += expf(x[c] - mx);
  float ls = logf(s);
  for (int c = 0; c < C; ++c) out[(size_t)r * C + c] = x[c] - mx - ls;
}

__global__ void copy_kernel(const float* __restrict__ in,
                            float* __restrict__ out, int n) {
  int t = blockIdx.x * blockDim.x + threadIdx.x;
  if (t < n) out[t] = in[t];
}

__global__ void zero_kernel(float* __restrict__ p, int n) {
  int t = blockIdx.x * blockDim.x + threadIdx.x;
  if (t < n) p[t] = 0.0f;
}

// ---------------------------------------------------------------------------
// Host orchestration
// ---------------------------------------------------------------------------
static inline void gemm(const bf16_t* A, int lda, int Mcompute, int Mstore,
                        int Kreal, int Kpad, int N, int ldo,
                        const float* const* L /*W,b,beta,g*/, void* out,
                        int outF32, bf16_t* wpack, int fuse, hipStream_t s) {
  int KC = Kpad / 32;
  int NT = (N + 15) / 16;
  int NTpad = (NT + 3) & ~3;
  int tot = NTpad * KC * 32;
  pack_w_kernel<<<(tot + 255) / 256, 256, 0, s>>>(L[0], Kreal, N, KC, NTpad, wpack);
  dim3 g((unsigned)((Mcompute / 16 + 7) / 8), (unsigned)(NTpad / 4));
  wmma_gemm_bf16<<<g, 256, 0, s>>>(A, lda, (const v16bf*)wpack, L[1], L[3], L[2],
                                   out, ldo, Mcompute, Mstore, KC, N, fuse, outF32);
}

extern "C" void kernel_launch(void* const* d_in, const int* in_sizes, int n_in,
                              void* d_out, int out_size, void* d_ws, size_t ws_size,
                              hipStream_t stream) {
  (void)in_sizes; (void)n_in; (void)out_size; (void)ws_size;

  const float* xyz = (const float*)d_in[0];

  // Params flattened after xyz: dict keys sorted -> heads(cl,d,n,r), sa1, sa2, sa3;
  // per linear layer keys sorted -> W, b, beta, g.
  int pi = 1;
  const float* headP[4][3][4];
  for (int h = 0; h < 4; ++h)
    for (int l = 0; l < 3; ++l)
      for (int a = 0; a < 4; ++a) headP[h][l][a] = (const float*)d_in[pi++];
  const float* sa1P[3][3][4];
  for (int m = 0; m < 3; ++m)
    for (int l = 0; l < 3; ++l)
      for (int a = 0; a < 4; ++a) sa1P[m][l][a] = (const float*)d_in[pi++];
  const float* sa2P[3][3][4];
  for (int m = 0; m < 3; ++m)
    for (int l = 0; l < 3; ++l)
      for (int a = 0; a < 4; ++a) sa2P[m][l][a] = (const float*)d_in[pi++];
  const float* sa3P[3][4];
  for (int l = 0; l < 3; ++l)
    for (int a = 0; a < 4; ++a) sa3P[l][a] = (const float*)d_in[pi++];

  // ---- workspace bump allocator ----
  char* base = (char*)d_ws;
  size_t off = 0;
  auto alloc = [&](size_t nbytes) -> void* {
    off = (off + 255) & ~(size_t)255;
    void* p = base + off;
    off += nbytes;
    return p;
  };

  float*  coords = (float*)alloc((size_t)BATCH * NPTS * 3 * 4);
  float*  norms  = (float*)alloc((size_t)BATCH * NPTS * 3 * 4);
  int*    fidx1  = (int*)  alloc((size_t)BATCH * 512 * 4);
  float*  nx1    = (float*)alloc((size_t)BATCH * 512 * 3 * 4);
  float*  l1pts  = (float*)alloc((size_t)BATCH * 512 * 320 * 4);
  int*    fidx2  = (int*)  alloc((size_t)BATCH * 128 * 4);
  float*  nx2    = (float*)alloc((size_t)BATCH * 128 * 3 * 4);
  float*  l2pts  = (float*)alloc((size_t)BATCH * 128 * 640 * 4);
  int*    idxbuf = (int*)  alloc((size_t)BATCH * 512 * 128 * 4);
  // bf16 ping-pong GEMM buffers (worst cases):
  //   bufA: max(SA2 grouped 131072x352, SA1-sc2 L1 524288x96 = 50.3M elems)
  //   bufB: SA1-sc2 L2 524288x128 = 67.1M elems
  bf16_t* bufA  = (bf16_t*)alloc((size_t)50331648 * 2);
  bf16_t* bufB  = (bf16_t*)alloc((size_t)67108864 * 2);
  bf16_t* wpack = (bf16_t*)alloc((size_t)2 * 1024 * 1024 * 2);
  float*  featF = (float*)alloc((size_t)16 * 1024 * 4);   // 16 rows (8 pad)
  bf16_t* featB = (bf16_t*)alloc((size_t)16 * 1024 * 2);
  bf16_t* h1    = (bf16_t*)alloc((size_t)16 * 512 * 2);
  bf16_t* h2    = (bf16_t*)alloc((size_t)16 * 256 * 2);
  float*  clraw = (float*)alloc((size_t)BATCH * 40 * 4);

  const int T = 256;
  split_xyz_kernel<<<(BATCH * NPTS + T - 1) / T, T, 0, stream>>>(xyz, coords, norms, NPTS);

  // ---------------- SA1 ----------------
  fps_kernel<<<BATCH, 256, 0, stream>>>(coords, NPTS, 512, fidx1);
  gather_xyz_kernel<<<(BATCH * 512 + T - 1) / T, T, 0, stream>>>(coords, fidx1, nx1, NPTS, 512);
  {
    const float radii[3] = {0.1f, 0.2f, 0.4f};
    const int ns[3] = {16, 32, 128};
    const int ch[3][3] = {{32, 32, 64}, {64, 64, 128}, {64, 96, 128}};
    int coff = 0;
    for (int sc = 0; sc < 3; ++sc) {
      int S = 512, k = ns[sc];
      int rows = BATCH * S * k;
      query_ball_kernel<<<(BATCH * S + T - 1) / T, T, 0, stream>>>(
          coords, nx1, NPTS, S, k, radii[sc] * radii[sc], idxbuf);
      group_kernel<<<(rows + T - 1) / T, T, 0, stream>>>(
          norms, 3, coords, nx1, idxbuf, S, k, NPTS, 32, bufA, rows);
      gemm(bufA, 32, rows, rows, 6, 32, ch[sc][0], ch[sc][0],
           sa1P[sc][0], bufB, 0, wpack, 1, stream);
      gemm(bufB, ch[sc][0], rows, rows, ch[sc][0], ch[sc][0], ch[sc][1],
           ch[sc][1], sa1P[sc][1], bufA, 0, wpack, 1, stream);
      gemm(bufA, ch[sc][1], rows, rows, ch[sc][1], ch[sc][1], ch[sc][2],
           ch[sc][2], sa1P[sc][2], bufB, 0, wpack, 1, stream);
      maxpool_kernel<<<((BATCH * S * ch[sc][2]) + T - 1) / T, T, 0, stream>>>(
          bufB, ch[sc][2], l1pts, BATCH * S, k, ch[sc][2], 320, coff);
      coff += ch[sc][2];
    }
  }

  // ---------------- SA2 ----------------
  fps_kernel<<<BATCH, 256, 0, stream>>>(nx1, 512, 128, fidx2);
  gather_xyz_kernel<<<(BATCH * 128 + T - 1) / T, T, 0, stream>>>(nx1, fidx2, nx2, 512, 128);
  {
    const float radii[3] = {0.2f, 0.4f, 0.8f};
    const int ns[3] = {32, 64, 128};
    const int ch[3][3] = {{64, 64, 128}, {128, 128, 256}, {128, 128, 256}};
    int coff = 0;
    for (int sc = 0; sc < 3; ++sc) {
      int S = 128, k = ns[sc];
      int rows = BATCH * S * k;
      query_ball_kernel<<<(BATCH * S + T - 1) / T, T, 0, stream>>>(
          nx1, nx2, 512, S, k, radii[sc] * radii[sc], idxbuf);
      group_kernel<<<(rows + T - 1) / T, T, 0, stream>>>(
          l1pts, 320, nx1, nx2, idxbuf, S, k, 512, 352, bufA, rows);
      gemm(bufA, 352, rows, rows, 323, 352, ch[sc][0], ch[sc][0],
           sa2P[sc][0], bufB, 0, wpack, 1, stream);
      gemm(bufB, ch[sc][0], rows, rows, ch[sc][0], ch[sc][0], ch[sc][1],
           ch[sc][1], sa2P[sc][1], bufA, 0, wpack, 1, stream);
      gemm(bufA, ch[sc][1], rows, rows, ch[sc][1], ch[sc][1], ch[sc][2],
           ch[sc][2], sa2P[sc][2], bufB, 0, wpack, 1, stream);
      maxpool_kernel<<<((BATCH * S * ch[sc][2]) + T - 1) / T, T, 0, stream>>>(
          bufB, ch[sc][2], l2pts, BATCH * S, k, ch[sc][2], 640, coff);
      coff += ch[sc][2];
    }
  }

  // ---------------- SA3 (global) ----------------
  sa3_concat_kernel<<<(BATCH * 128 + T - 1) / T, T, 0, stream>>>(nx2, l2pts, bufA, 672);
  {
    int rows = BATCH * 128;
    gemm(bufA, 672, rows, rows, 643, 672, 256, 256, sa3P[0], bufB, 0, wpack, 1, stream);
    gemm(bufB, 256, rows, rows, 256, 256, 512, 512, sa3P[1], bufA, 0, wpack, 1, stream);
    gemm(bufA, 512, rows, rows, 512, 512, 1024, 1024, sa3P[2], bufB, 0, wpack, 1, stream);
    maxpool_kernel<<<((BATCH * 1024) + T - 1) / T, T, 0, stream>>>(
        bufB, 1024, featF, BATCH, 128, 1024, 1024, 0);
    zero_kernel<<<(8 * 1024 + T - 1) / T, T, 0, stream>>>(featF + 8 * 1024, 8 * 1024);
    f32_to_bf16_kernel<<<(16 * 1024 + T - 1) / T, T, 0, stream>>>(featF, featB, 16 * 1024);
  }

  // ---------------- Heads ----------------
  float* outF = (float*)d_out;
  float* cl_out   = outF;                 // 8*40
  float* d_outp   = cl_out + BATCH * 40;  // 8*3
  float* n_outp   = d_outp + BATCH * 3;   // 8*3
  float* r_outp   = n_outp + BATCH * 3;   // 8*1
  float* feat_out = r_outp + BATCH * 1;   // 8*1024
  const int hdo[4] = {40, 3, 3, 1};
  float* hdst[4] = {clraw, d_outp, n_outp, r_outp};
  for (int h = 0; h < 4; ++h) {
    gemm(featB, 1024, 16, 16, 1024, 1024, 512, 512, headP[h][0], h1, 0, wpack, 1, stream);
    gemm(h1, 512, 16, 16, 512, 512, 256, 256, headP[h][1], h2, 0, wpack, 1, stream);
    gemm(h2, 256, 16, BATCH, 256, 256, hdo[h], hdo[h], headP[h][2],
         hdst[h], 1, wpack, 0, stream);
  }
  log_softmax_kernel<<<1, 32, 0, stream>>>(clraw, cl_out, BATCH, 40);
  copy_kernel<<<(BATCH * 1024 + T - 1) / T, T, 0, stream>>>(featF, feat_out, BATCH * 1024);
}
